// MOC_Branch_58935541236380
// MI455X (gfx1250) — matrix operand order repaired and checked by reading, not compiled, not run on hardware
//
#include <hip/hip_runtime.h>
#include <hip/hip_bf16.h>

typedef __attribute__((ext_vector_type(16))) _Float16 v16h;
typedef __attribute__((ext_vector_type(8)))  float    v8f;
typedef __attribute__((ext_vector_type(4)))  int      v4i;
typedef unsigned int uint;

union FragAB { v16h h; uint u[8]; };

// Async global->LDS copy path (gfx1250): guarded so compile can't regress.
#if defined(__HIP_DEVICE_COMPILE__) && __has_builtin(__builtin_amdgcn_global_load_async_to_lds_b128) && __has_builtin(__builtin_amdgcn_s_wait_asynccnt)
#define HAVE_ASYNC_LDS 1
typedef __attribute__((address_space(1))) v4i glob_v4i;
typedef __attribute__((address_space(3))) v4i lds_v4i;
#else
#define HAVE_ASYNC_LDS 0
#endif

// A-matrix (16x32 f16) K placement per CDNA5 ISA:
// lanes 0-15: VGPR0-3 -> K0..7, VGPR4-7 -> K16..23 ; lanes 16-31: +8
__device__ __forceinline__ int kmapA(int v, int hi) {
    return ((v < 4) ? 2 * v : 2 * v + 8) + (hi ? 8 : 0);
}

// ---------------------------------------------------------------------------
// Kernel 1: features f32 (K,B,C,96,96) -> padded f16 P[b][k*64+c][98][98]
// ---------------------------------------------------------------------------
__global__ void build_pad_kernel(const float* __restrict__ feat, _Float16* __restrict__ P) {
    size_t idx = (size_t)blockIdx.x * 256 + threadIdx.x;
    const size_t TOT = (size_t)4 * 448 * 9604;
    if (idx >= TOT) return;
    int p = (int)(idx % 9604);
    size_t rest = idx / 9604;
    int kc = (int)(rest % 448);
    int b  = (int)(rest / 448);
    int yp = p / 98, xp = p - yp * 98;
    float val = 0.f;
    if (yp >= 1 && yp <= 96 && xp >= 1 && xp <= 96) {
        int k = kc >> 6, c = kc & 63;
        val = feat[((size_t)((k * 4 + b) * 64 + c)) * 9216 + (size_t)(yp - 1) * 96 + (xp - 1)];
    }
    P[idx] = (_Float16)val;
}

// ---------------------------------------------------------------------------
// Kernel 2: f32 weight (Msrc x K) -> f16 (Mdst x K), zero-pad extra rows
// ---------------------------------------------------------------------------
__global__ void convert_rows_kernel(const float* __restrict__ src, _Float16* __restrict__ dst,
                                    int Msrc, int K, long total) {
    long idx = (long)blockIdx.x * 256 + threadIdx.x;
    if (idx >= total) return;
    int m = (int)(idx / K);
    dst[idx] = (m < Msrc) ? (_Float16)src[idx] : (_Float16)0.f;
}

// ---------------------------------------------------------------------------
// Kernel 3: small constants: ones, padded off bias, folded BN scale/shift
// ---------------------------------------------------------------------------
__global__ void misc_init_kernel(float* ones, float* offshift, float* bnscale, float* bnshift,
                                 const float* __restrict__ off_b, const float* __restrict__ gamma,
                                 const float* __restrict__ beta, const float* __restrict__ mean,
                                 const float* __restrict__ var, const float* __restrict__ dcn_b) {
    int t = threadIdx.x;
    ones[t] = 1.f;
    offshift[t] = (t < 189) ? off_b[t] : 0.f;
    float s = gamma[t] * rsqrtf(var[t] + 1e-5f);
    bnscale[t] = s;
    bnshift[t] = (dcn_b[t] - mean[t]) * s + beta[t];
}

// ---------------------------------------------------------------------------
// Kernel 4: fused conv3x3(256ch WMMA GEMM) -> scale/shift(+ReLU) -> 1x1 WMMA
// heads: 0=wh 1=hm 2=id 3=off(raw f32, M2=0) 4=dcn(cols direct B, BN folded)
// block = 256 threads = 8 waves; grid = (96 rows, n_images)
// A tile (256x32 f16, 16KB) staged in LDS (async copy), aliased with mid tile.
// ---------------------------------------------------------------------------
__global__ __launch_bounds__(256) void fused_head_kernel(
    int head,
    const _Float16* __restrict__ A1, int Ktot,
    const float* __restrict__ scale1, const float* __restrict__ shift1, int relu1,
    const _Float16* __restrict__ A2, const float* __restrict__ b2, int M2,
    const _Float16* __restrict__ Bsrc,
    float* __restrict__ outBase, int imgArg)
{
    __shared__ _Float16 sAM[96 * 256];   // k-loop: A tile [m][k] (first 8192) ; after: mid [px][ch]
    __shared__ _Float16 sB[96 * 32];     // B tile [px][k]   6 KB
    __shared__ float sScale[256];
    __shared__ float sShift[256];
    __shared__ float sB2[128];

    const int t    = threadIdx.x;
    const int lane = t & 31;
    const int wave = t >> 5;
    const int lm   = lane & 15;
    const int hi   = lane >> 4;
    const int y    = blockIdx.x;
    const int img  = blockIdx.y;

    const _Float16* src;
    float* out;
    if (head == 0) {            // wh: 28 images (k,b)
        int k = img >> 2, bb = img & 3;
        src = Bsrc + (size_t)(bb * 448 + k * 64) * 9604;
        out = outBase + (size_t)(bb * 14 + k * 2) * 9216;
    } else if (head == 1) {     // hm: features[3]
        src = Bsrc + (size_t)(img * 448 + 192) * 9604;
        out = outBase + (size_t)img * 24 * 9216;
    } else if (head == 2) {     // id: cat
        src = Bsrc + (size_t)img * 448 * 9604;
        out = outBase + (size_t)img * 128 * 9216;
    } else if (head == 3) {     // off conv, raw f32 out (256-ch stride)
        src = Bsrc + (size_t)img * 448 * 9604;
        out = outBase + (size_t)img * 256 * 9216;
    } else {                    // dcn: cols already per-batch
        src = Bsrc;
        out = outBase + (size_t)imgArg * 16 * 9216;
    }

    sScale[t] = scale1[t];
    sShift[t] = shift1[t];
    if (b2 && t < M2) sB2[t] = b2[t];

    v8f zero = {0.f, 0.f, 0.f, 0.f, 0.f, 0.f, 0.f, 0.f};
    v8f acc[2][6];
#pragma unroll
    for (int i = 0; i < 2; ++i)
#pragma unroll
        for (int j = 0; j < 6; ++j) acc[i][j] = zero;

    const int kk   = t >> 3;  // 0..31 : K row within chunk (B loader)
    const int seg  = t & 7;   // 0..7  : 12-pixel segment   (B loader)
    const int am   = t >> 2;  // 0..63 : A row group        (A loader)
    const int part = t & 3;   // 0..3  : 16B part of 64B row (A loader)

    for (int k0 = 0; k0 < Ktot; k0 += 32) {
        __syncthreads();
        // ---- stage A chunk (256 rows x 32 k) into LDS, coalesced 16B/lane
#pragma unroll
        for (int r = 0; r < 4; ++r) {
            int m = r * 64 + am;
            const _Float16* gsrc = A1 + (size_t)m * Ktot + k0 + part * 8;
            _Float16* ldst = &sAM[m * 32 + part * 8];
#if HAVE_ASYNC_LDS
            __builtin_amdgcn_global_load_async_to_lds_b128(
                (glob_v4i*)gsrc, (lds_v4i*)ldst, 0, 0);
#else
            *(uint4*)ldst = *(const uint4*)gsrc;
#endif
        }
        // ---- stage B chunk (32 k x 96 px) into LDS, [px][k] layout
        int krow = k0 + kk;
        if (head < 4) {  // im2col on the fly from padded image
            int c = krow / 9;
            int r = krow - c * 9;
            int ky = r / 3;
            int kx = r - ky * 3;
            const _Float16* rowp = src + (size_t)c * 9604 + (size_t)(y + ky) * 98 + kx;
#pragma unroll
            for (int j = 0; j < 12; ++j) {
                int px = seg * 12 + j;
                sB[px * 32 + kk] = rowp[px];
            }
        } else {         // direct rows from cols matrix
            const _Float16* rowp = src + (size_t)krow * 9216 + (size_t)y * 96;
#pragma unroll
            for (int j = 0; j < 12; ++j) {
                int px = seg * 12 + j;
                sB[px * 32 + kk] = rowp[px];
            }
        }
#if HAVE_ASYNC_LDS
        __builtin_amdgcn_s_wait_asynccnt(0);
#endif
        __syncthreads();

        FragAB afrag[2], bfrag[6];
#pragma unroll
        for (int i = 0; i < 2; ++i) {
            const _Float16* arow = &sAM[((wave * 2 + i) * 16 + lm) * 32];
#pragma unroll
            for (int v = 0; v < 8; ++v)
                afrag[i].u[v] = *(const uint*)(arow + kmapA(v, hi));
        }
#pragma unroll
        for (int ct = 0; ct < 6; ++ct) {
            const _Float16* brow = &sB[(ct * 16 + lm) * 32];
#pragma unroll
            for (int v = 0; v < 8; ++v)
                bfrag[ct].u[v] = *(const uint*)(brow + kmapA(v, hi));
        }
#pragma unroll
        for (int i = 0; i < 2; ++i)
#pragma unroll
            for (int ct = 0; ct < 6; ++ct)
                acc[i][ct] = __builtin_amdgcn_wmma_f32_16x16x32_f16(
                    false, afrag[i].h, false, bfrag[ct].h, (short)0, acc[i][ct], false, false);
        if (k0 + 32 < Ktot)
            __builtin_prefetch((const void*)(A1 + (size_t)t * Ktot + k0 + 32), 0, 0);
    }
    __syncthreads();  // all waves done reading sAM-as-A before it becomes sMid

    // epilogue 1: scale/shift (+relu); either write f32 out or stage mid f16
#pragma unroll
    for (int i = 0; i < 2; ++i) {
        int rt = wave * 2 + i;
#pragma unroll
        for (int ct = 0; ct < 6; ++ct) {
            int px = ct * 16 + lm;
#pragma unroll
            for (int v = 0; v < 8; ++v) {
                int M = rt * 16 + v + hi * 8;
                float val = acc[i][ct][v] * sScale[M] + sShift[M];
                if (relu1) val = fmaxf(val, 0.f);
                if (M2 == 0)
                    out[(size_t)M * 9216 + (size_t)y * 96 + px] = val;
                else
                    sAM[px * 256 + M] = (_Float16)val;   // sMid
            }
        }
    }
    if (M2 == 0) return;
    __syncthreads();

    // second GEMM: (M2 x 256) x (256 x 96) from LDS mid tile
    int nrt2 = (M2 + 15) >> 4;
    int T2 = nrt2 * 6;
    for (int tt = wave; tt < T2; tt += 8) {
        int rt2 = tt / 6;
        int ct  = tt - rt2 * 6;
        int px  = ct * 16 + lm;
        v8f acc2 = zero;
#pragma unroll
        for (int ks = 0; ks < 8; ++ks) {
            int k2 = ks * 32;
            FragAB af, bf;
            const _Float16* arow = A2 + (size_t)(rt2 * 16 + lm) * 256 + k2;
#pragma unroll
            for (int v = 0; v < 8; ++v) af.u[v] = *(const uint*)(arow + kmapA(v, hi));
            const _Float16* brow = &sAM[px * 256 + k2];
#pragma unroll
            for (int v = 0; v < 8; ++v) bf.u[v] = *(const uint*)(brow + kmapA(v, hi));
            acc2 = __builtin_amdgcn_wmma_f32_16x16x32_f16(
                false, af.h, false, bf.h, (short)0, acc2, false, false);
        }
#pragma unroll
        for (int v = 0; v < 8; ++v) {
            int o2 = rt2 * 16 + v + hi * 8;
            if (o2 < M2)
                out[(size_t)o2 * 9216 + (size_t)y * 96 + px] = acc2[v] + sB2[o2];
        }
    }
}

// ---------------------------------------------------------------------------
// Kernel 5: DCNv2 sampling -> cols f16 matrix (rows (g*64+c)*9+kk, cols pix)
// one thread per (g,kk,pix), loops the 64 group channels
// ---------------------------------------------------------------------------
__global__ __launch_bounds__(256) void dcn_sample_kernel(
    const float* __restrict__ feat, const float* __restrict__ om,
    _Float16* __restrict__ cols, int b)
{
    int tid = blockIdx.x * 256 + threadIdx.x;
    const int TOT = 7 * 9 * 9216;
    if (tid >= TOT) return;
    int g   = tid / (9 * 9216);
    int rem = tid - g * (9 * 9216);
    int kk  = rem / 9216;
    int pix = rem - kk * 9216;
    int yy = pix / 96, xx = pix - yy * 96;
    int ky = kk / 3,   kx = kk - ky * 3;
    int ch = g * 9 + kk;
    float offY = om[(size_t)ch * 9216 + pix];
    float offX = om[(size_t)(63 + ch) * 9216 + pix];
    float mraw = om[(size_t)(126 + ch) * 9216 + pix];
    float mk = 1.f / (1.f + expf(-mraw));
    float sy = (float)(yy - 1 + ky) + offY;
    float sx = (float)(xx - 1 + kx) + offX;
    float y0f = floorf(sy), x0f = floorf(sx);
    float wy = sy - y0f, wx = sx - x0f;
    int y0 = (int)y0f, x0 = (int)x0f;
    int y1 = y0 + 1,   x1 = x0 + 1;
    float v00 = (y0 >= 0 && y0 < 96 && x0 >= 0 && x0 < 96) ? 1.f : 0.f;
    float v01 = (y0 >= 0 && y0 < 96 && x1 >= 0 && x1 < 96) ? 1.f : 0.f;
    float v10 = (y1 >= 0 && y1 < 96 && x0 >= 0 && x0 < 96) ? 1.f : 0.f;
    float v11 = (y1 >= 0 && y1 < 96 && x1 >= 0 && x1 < 96) ? 1.f : 0.f;
    int cy0 = y0 < 0 ? 0 : (y0 > 95 ? 95 : y0);
    int cx0 = x0 < 0 ? 0 : (x0 > 95 ? 95 : x0);
    int cy1 = y1 < 0 ? 0 : (y1 > 95 ? 95 : y1);
    int cx1 = x1 < 0 ? 0 : (x1 > 95 ? 95 : x1);
    int i00 = cy0 * 96 + cx0, i01 = cy0 * 96 + cx1;
    int i10 = cy1 * 96 + cx0, i11 = cy1 * 96 + cx1;
    float w00 = (1.f - wy) * (1.f - wx) * v00 * mk;
    float w01 = (1.f - wy) * wx * v01 * mk;
    float w10 = wy * (1.f - wx) * v10 * mk;
    float w11 = wy * wx * v11 * mk;
    const float* X = feat + (size_t)(g * 4 + b) * 64 * 9216;
    _Float16* crow = cols + ((size_t)(g * 64) * 9 + kk) * 9216 + pix;
#pragma unroll 4
    for (int c = 0; c < 64; ++c) {
        const float* Xc = X + (size_t)c * 9216;
        float v = w00 * Xc[i00] + w01 * Xc[i01] + w10 * Xc[i10] + w11 * Xc[i11];
        crow[(size_t)c * 9 * 9216] = (_Float16)v;
    }
}

// ---------------------------------------------------------------------------
extern "C" void kernel_launch(void* const* d_in, const int* in_sizes, int n_in,
                              void* d_out, int out_size, void* d_ws, size_t ws_size,
                              hipStream_t stream) {
    (void)in_sizes; (void)n_in; (void)out_size; (void)ws_size;
    const float* features = (const float*)d_in[0];
    const float* hm_w1 = (const float*)d_in[1];
    const float* hm_b1 = (const float*)d_in[2];
    const float* hm_w2 = (const float*)d_in[3];
    const float* hm_b2 = (const float*)d_in[4];
    const float* wh_w1 = (const float*)d_in[5];
    const float* wh_b1 = (const float*)d_in[6];
    const float* wh_w2 = (const float*)d_in[7];
    const float* wh_b2 = (const float*)d_in[8];
    const float* id_w1 = (const float*)d_in[9];
    const float* id_b1 = (const float*)d_in[10];
    const float* id_w2 = (const float*)d_in[11];
    const float* id_b2 = (const float*)d_in[12];
    const float* off_w = (const float*)d_in[13];
    const float* off_b = (const float*)d_in[14];
    const float* dcn_w = (const float*)d_in[15];
    const float* dcn_b = (const float*)d_in[16];
    const float* bn_gamma = (const float*)d_in[17];
    const float* bn_beta  = (const float*)d_in[18];
    const float* bn_mean  = (const float*)d_in[19];
    const float* bn_var   = (const float*)d_in[20];
    const float* bz_w = (const float*)d_in[21];
    const float* bz_b = (const float*)d_in[22];

    char* ws = (char*)d_ws;
    size_t cur = 0;
    auto take = [&](size_t bytes) -> void* {
        void* p = ws + cur;
        cur = (cur + bytes + 255) & ~(size_t)255;
        return p;
    };
    _Float16* P     = (_Float16*)take((size_t)4 * 448 * 9604 * 2);  // 34.4 MB
    _Float16* A_wh  = (_Float16*)take((size_t)256 * 576 * 2);
    _Float16* A_hm  = (_Float16*)take((size_t)256 * 576 * 2);
    _Float16* A_id  = (_Float16*)take((size_t)256 * 4032 * 2);
    _Float16* A_off = (_Float16*)take((size_t)256 * 4032 * 2);
    _Float16* A_dcn = (_Float16*)take((size_t)256 * 4032 * 2);
    _Float16* A2_wh = (_Float16*)take((size_t)16  * 256 * 2);
    _Float16* A2_hm = (_Float16*)take((size_t)32  * 256 * 2);
    _Float16* A2_id = (_Float16*)take((size_t)128 * 256 * 2);
    _Float16* A2_bz = (_Float16*)take((size_t)16  * 256 * 2);
    float* ones     = (float*)take(256 * 4);
    float* offshift = (float*)take(256 * 4);
    float* bnscale  = (float*)take(256 * 4);
    float* bnshift  = (float*)take(256 * 4);
    float* om       = (float*)take((size_t)4 * 256 * 9216 * 4);     // 37.7 MB
    _Float16* cols  = (_Float16*)take((size_t)4032 * 9216 * 2);     // 74.3 MB

    float* hm_out  = (float*)d_out;
    float* bz_out  = hm_out + (size_t)4 * 24 * 9216;
    float* id_out  = bz_out + (size_t)4 * 16 * 9216;
    float* owh_out = id_out + (size_t)4 * 128 * 9216;

    dim3 blk(256);

    {   // padded f16 features
        size_t tot = (size_t)4 * 448 * 9604;
        build_pad_kernel<<<dim3((unsigned)((tot + 255) / 256)), blk, 0, stream>>>(features, P);
    }
    auto conv = [&](const float* s, _Float16* d, int Msrc, int K, int Mdst) {
        long tot = (long)Mdst * K;
        convert_rows_kernel<<<dim3((unsigned)((tot + 255) / 256)), blk, 0, stream>>>(s, d, Msrc, K, tot);
    };
    conv(wh_w1, A_wh, 256, 576, 256);
    conv(hm_w1, A_hm, 256, 576, 256);
    conv(id_w1, A_id, 256, 4032, 256);
    conv(off_w, A_off, 189, 4032, 256);
    conv(dcn_w, A_dcn, 256, 4032, 256);
    conv(wh_w2, A2_wh, 2, 256, 16);
    conv(hm_w2, A2_hm, 24, 256, 32);
    conv(id_w2, A2_id, 128, 256, 128);
    conv(bz_w,  A2_bz, 16, 256, 16);
    misc_init_kernel<<<dim3(1), blk, 0, stream>>>(ones, offshift, bnscale, bnshift,
                                                  off_b, bn_gamma, bn_beta, bn_mean, bn_var, dcn_b);

    // wh head: 28 images, conv3x3(64->256)+relu -> 1x1(256->2)
    fused_head_kernel<<<dim3(96, 28), blk, 0, stream>>>(
        0, A_wh, 576, ones, wh_b1, 1, A2_wh, wh_b2, 2, P, owh_out, 0);
    // hm head: features[3], 1x1(256->24)
    fused_head_kernel<<<dim3(96, 4), blk, 0, stream>>>(
        1, A_hm, 576, ones, hm_b1, 1, A2_hm, hm_b2, 24, P, hm_out, 0);
    // id head: cat, conv3x3(448->256)+relu -> 1x1(256->128)
    fused_head_kernel<<<dim3(96, 4), blk, 0, stream>>>(
        2, A_id, 4032, ones, id_b1, 1, A2_id, id_b2, 128, P, id_out, 0);
    // offset/mask conv: raw f32, no relu, no second gemm
    fused_head_kernel<<<dim3(96, 4), blk, 0, stream>>>(
        3, A_off, 4032, ones, offshift, 0, nullptr, nullptr, 0, P, om, 0);
    // dcn per batch: sample -> grouped GEMM (BN folded) + relu -> bz 1x1
    for (int b = 0; b < 4; ++b) {
        dcn_sample_kernel<<<dim3(2268), blk, 0, stream>>>(
            features, om + (size_t)b * 256 * 9216, cols, b);
        fused_head_kernel<<<dim3(96, 1), blk, 0, stream>>>(
            4, A_dcn, 4032, bnscale, bnshift, 1, A2_bz, bz_b, 16, cols, bz_out, b);
    }
}
